// RIMNetwork_multHead_68736656605805
// MI455X (gfx1250) — matrix operand order repaired and checked by reading, not compiled, not run on hardware
//
#include <hip/hip_runtime.h>
#include <hip/hip_bf16.h>

#define Bv 128
#define Tv 256
#define Iv 128
#define Hv 256
#define Mv 8
#define Kv 32
#define Rv 16
#define Ov 128
#define Dv 256

typedef __attribute__((ext_vector_type(16))) __bf16 bf16x16;
typedef __attribute__((ext_vector_type(8)))  float  f32x8;
typedef __attribute__((ext_vector_type(4)))  float  f32x4;

// ---------- bf16 helpers (native converts: RNE fptrunc/fpext -> v_cvt ops) ----------
__device__ inline __bf16 bf_zero() {
  unsigned short z = 0; __bf16 r; __builtin_memcpy(&r, &z, 2); return r;
}
__device__ inline __bf16 f2bf(float f) { return (__bf16)f; }
__device__ inline float bf2f(__bf16 h) { return (float)h; }

// k-position swizzle inside each 32-wide K window: swap bits 3 and 4.
// Storage order [k0..7, k16..23, k8..15, k24..31] makes each lane-half's
// A-fragment (CDNA5 16-bit 16x32 layout) a contiguous 16-element run.
__device__ inline int swz32(int h) {
  return (h & ~24) | ((h & 8) << 1) | ((h & 16) >> 1);
}

// ---------- fragment loaders ----------
// A from swizzled bf16 LDS row: one contiguous 32B read per lane.
__device__ inline bf16x16 load_a_lds(const __bf16* rowbase_k) {
  int half = (threadIdx.x >> 4) & 1;
  return *(const bf16x16*)(rowbase_k + 16 * half);
}
// A from row-major f32 global: two 8-float runs (4x float4) + convert.
__device__ inline bf16x16 load_a_f32v(const float* rowptr_k) {
  int half = (threadIdx.x >> 4) & 1;
  const float* p0 = rowptr_k + 8 * half;
  const float* p1 = rowptr_k + 16 + 8 * half;
  f32x4 r0 = *(const f32x4*)(p0);
  f32x4 r1 = *(const f32x4*)(p0 + 4);
  f32x4 r2 = *(const f32x4*)(p1);
  f32x4 r3 = *(const f32x4*)(p1 + 4);
  bf16x16 a;
#pragma unroll
  for (int e = 0; e < 4; ++e) {
    a[e]      = f2bf(r0[e]);
    a[4 + e]  = f2bf(r1[e]);
    a[8 + e]  = f2bf(r2[e]);
    a[12 + e] = f2bf(r3[e]);
  }
  return a;
}
// B from pre-swizzled weight tiles: tile = 512 bf16 (32 lanes x 16), one 32B read.
__device__ inline bf16x16 load_b_tile(const __bf16* mat, int tile) {
  int lane = threadIdx.x & 31;
  return *(const bf16x16*)(mat + ((long long)tile << 9) + (lane << 4));
}
__device__ inline f32x8 wmma_bf16(bf16x16 a, bf16x16 b, f32x8 c) {
  return __builtin_amdgcn_wmma_f32_16x16x32_bf16(false, a, false, b, (short)0, c,
                                                 false, false);
}

// ---------- workspace layout (bf16 weights, fragment-tiled) ----------
// Tile order per matrix noted below; each tile = 512 bf16.
#define O_WIN 0LL
#define N_WIN (8LL * 4 * 16 * 512)   // (m, kt4, nt16)
#define O_WP  (O_WIN + N_WIN)
#define N_WP  (8LL * 8 * 16 * 512)   // (m, kt8, nt16)
#define O_U   (O_WP + N_WP)
#define N_U   (8LL * 8 * 1 * 512)    // (m, kt8)
#define O_VR  (O_U + N_U)
#define N_VR  (8LL * 1 * 16 * 512)   // (m, nt16)  K padded 16->32 with zeros
#define O_WQ  (O_VR + N_VR)
#define N_WQ  (8LL * 8 * 2 * 512)    // (m, kt8, nt2)
#define O_WK  (O_WQ + N_WQ)
#define N_WK  (8LL * 8 * 2 * 512)
#define O_WV  (O_WK + N_WK)
#define N_WV  (8LL * 8 * 16 * 512)   // (m, kt8, nt16)
#define O_WOT (O_WV + N_WV)
#define N_WOT (64LL * 8 * 512)       // (kt64, nt8), transposed Wo
#define N_TOT (O_WOT + N_WOT)

// ---------- kernel 0: fp32 -> bf16 convert + fragment-tile swizzle ----------
__global__ void k_convert(const float* __restrict__ Win, const float* __restrict__ Wp,
                          const float* __restrict__ U, const float* __restrict__ Vr,
                          const float* __restrict__ WQ, const float* __restrict__ WK,
                          const float* __restrict__ WV, const float* __restrict__ Wo,
                          __bf16* __restrict__ ws) {
  long long i = (long long)blockIdx.x * blockDim.x + threadIdx.x;
  if (i >= N_TOT) return;
  float v;
  if (i < O_WP) {            // W_in [M][I][H]
    long long rel = i - O_WIN; int tile = (int)(rel >> 9), wi = (int)(rel & 511);
    int lane = wi >> 4, e = wi & 15, half = lane >> 4, col = lane & 15;
    int kit = (e & 7) + 8 * half + 16 * (e >> 3);
    int nt = tile & 15, kt = (tile >> 4) & 3, m = tile >> 6;
    v = Win[((long long)m * Iv + kt * 32 + kit) * Hv + nt * 16 + col];
  } else if (i < O_U) {      // Wp [M][H][H]
    long long rel = i - O_WP; int tile = (int)(rel >> 9), wi = (int)(rel & 511);
    int lane = wi >> 4, e = wi & 15, half = lane >> 4, col = lane & 15;
    int kit = (e & 7) + 8 * half + 16 * (e >> 3);
    int nt = tile & 15, kt = (tile >> 4) & 7, m = tile >> 7;
    v = Wp[((long long)m * Hv + kt * 32 + kit) * Hv + nt * 16 + col];
  } else if (i < O_VR) {     // U [M][H][R]
    long long rel = i - O_U; int tile = (int)(rel >> 9), wi = (int)(rel & 511);
    int lane = wi >> 4, e = wi & 15, half = lane >> 4, col = lane & 15;
    int kit = (e & 7) + 8 * half + 16 * (e >> 3);
    int kt = tile & 7, m = tile >> 3;
    v = U[((long long)m * Hv + kt * 32 + kit) * Rv + col];
  } else if (i < O_WQ) {     // Vr [M][R][H], K padded to 32
    long long rel = i - O_VR; int tile = (int)(rel >> 9), wi = (int)(rel & 511);
    int lane = wi >> 4, e = wi & 15, half = lane >> 4, col = lane & 15;
    int kit = (e & 7) + 8 * half + 16 * (e >> 3);
    int nt = tile & 15, m = tile >> 4;
    v = (kit < Rv) ? Vr[((long long)m * Rv + kit) * Hv + nt * 16 + col] : 0.f;
  } else if (i < O_WK) {     // W_Q [1][M][H][K]
    long long rel = i - O_WQ; int tile = (int)(rel >> 9), wi = (int)(rel & 511);
    int lane = wi >> 4, e = wi & 15, half = lane >> 4, col = lane & 15;
    int kit = (e & 7) + 8 * half + 16 * (e >> 3);
    int nt = tile & 1, kt = (tile >> 1) & 7, m = tile >> 4;
    v = WQ[((long long)m * Hv + kt * 32 + kit) * Kv + nt * 16 + col];
  } else if (i < O_WV) {     // W_K [1][M][H][K]
    long long rel = i - O_WK; int tile = (int)(rel >> 9), wi = (int)(rel & 511);
    int lane = wi >> 4, e = wi & 15, half = lane >> 4, col = lane & 15;
    int kit = (e & 7) + 8 * half + 16 * (e >> 3);
    int nt = tile & 1, kt = (tile >> 1) & 7, m = tile >> 4;
    v = WK[((long long)m * Hv + kt * 32 + kit) * Kv + nt * 16 + col];
  } else if (i < O_WOT) {    // W_V [M][H][D]
    long long rel = i - O_WV; int tile = (int)(rel >> 9), wi = (int)(rel & 511);
    int lane = wi >> 4, e = wi & 15, half = lane >> 4, col = lane & 15;
    int kit = (e & 7) + 8 * half + 16 * (e >> 3);
    int nt = tile & 15, kt = (tile >> 4) & 7, m = tile >> 7;
    v = WV[((long long)m * Hv + kt * 32 + kit) * Dv + nt * 16 + col];
  } else {                   // Wo [O][M*H] -> B layout [in][out]
    long long rel = i - O_WOT; int tile = (int)(rel >> 9), wi = (int)(rel & 511);
    int lane = wi >> 4, e = wi & 15, half = lane >> 4, col = lane & 15;
    int kit = (e & 7) + 8 * half + 16 * (e >> 3);
    int nt = tile & 7, kt = tile >> 3;
    v = Wo[((long long)(nt * 16 + col)) * 2048 + kt * 32 + kit];
  }
  ws[i] = f2bf(v);
}

// ---------- kernel 1: A_in[t,b,m,h] = x[b,t,:] @ W_in[m] ----------
__global__ __launch_bounds__(32) void k_ain(const float* __restrict__ x,
                                            const __bf16* __restrict__ Winb,
                                            float* __restrict__ ain) {
  int rt = blockIdx.x, ct = blockIdx.y;  // 2048 x 128
  int m = ct >> 4, ht = ct & 15;
  int lane = threadIdx.x & 31; int half = lane >> 4; int lr = lane & 15;
  int row = rt * 16 + lr; int t = row >> 7; int b = row & 127;  // row = t*B + b
  const float* rowptr = x + ((long long)b * Tv + t) * Iv;
  f32x8 c = {};
  for (int kt = 0; kt < 4; ++kt) {
    bf16x16 a  = load_a_f32v(rowptr + kt * 32);
    bf16x16 bb = load_b_tile(Winb, (m * 4 + kt) * 16 + ht);
    c = wmma_bf16(a, bb, c);
  }
#pragma unroll
  for (int i = 0; i < 8; ++i) {
    int rr = rt * 16 + i + 8 * half;
    int tt = rr >> 7, b2 = rr & 127;
    ain[(((long long)tt * Bv + b2) * Mv + m) * Hv + ht * 16 + lr] = c[i];
  }
}

// ---------- kernel 2: ip[b,t,m,g] = A_in@Wp + bp, staged into attention_comm ----------
__global__ __launch_bounds__(32) void k_ip(const float* __restrict__ ain,
                                           const __bf16* __restrict__ Wpb,
                                           const float* __restrict__ bp,
                                           float* __restrict__ ip) {
  int rt = blockIdx.x, ct = blockIdx.y;
  int m = ct >> 4, ht = ct & 15;
  int lane = threadIdx.x & 31; int half = lane >> 4; int lr = lane & 15;
  int row = rt * 16 + lr; int b = row >> 8; int t = row & 255;  // row = b*T + t
  const float* rowptr = ain + (((long long)t * Bv + b) * Mv + m) * Hv;
  f32x8 c = {};
  for (int kt = 0; kt < 8; ++kt) {
    bf16x16 a  = load_a_f32v(rowptr + kt * 32);
    bf16x16 bb = load_b_tile(Wpb, (m * 8 + kt) * 16 + ht);
    c = wmma_bf16(a, bb, c);
  }
  float bias = bp[m * Hv + ht * 16 + lr];
#pragma unroll
  for (int i = 0; i < 8; ++i) {
    int rr = rt * 16 + i + 8 * half;
    int b2 = rr >> 8, t2 = rr & 255;
    ip[(((long long)b2 * Tv + t2) * Mv + m) * Hv + ht * 16 + lr] = c[i] + bias;
  }
}

// ---------- kernel 3: recurrence; 8 blocks x 16 waves, T loop in-kernel ----------
__global__ __launch_bounds__(512) void k_recurrent(
    const float* __restrict__ bias_m, const __bf16* __restrict__ Ub,
    const __bf16* __restrict__ Vrb, const __bf16* __restrict__ WQb,
    const __bf16* __restrict__ WKb, const __bf16* __restrict__ WVb,
    float* __restrict__ hidden /*[T,B,M,H]*/,
    float* __restrict__ comm /*[B,T,M,H], ip in, A_comm out*/,
    float* __restrict__ attw /*[1,T,B,M,M]*/) {
  __shared__ __align__(32) __bf16 Hs[16][Mv * Hv];  // 64KB, k-swizzled carry
  __shared__ __align__(32) __bf16 Sb[Mv][16][32];   // 8KB, low-rank, K padded
  __shared__ float Qb[16][Mv][Kv];                  // 16KB
  __shared__ float Kb[16][Mv][Kv];                  // 16KB
  __shared__ float Wb[16][Mv][Mv];                  // 4KB, smoothed attn carry

  const int tid = threadIdx.x;
  const int wid = tid >> 5;  // 16 waves
  const int lane = tid & 31;
  const int half = lane >> 4;
  const int lr = lane & 15;
  const int b0 = blockIdx.x * 16;

  for (int i = tid; i < 16 * Mv * Hv; i += 512) ((__bf16*)Hs)[i] = bf_zero();
  for (int i = tid; i < Mv * 16 * 32; i += 512) ((__bf16*)Sb)[i] = bf_zero();
  for (int i = tid; i < 16 * Mv * Mv; i += 512) ((float*)Wb)[i] = 0.f;
  __syncthreads();

  for (int t = 0; t < Tv; ++t) {
    // Prefetch next step's ip cachelines (gfx1250 global_prefetch_b8):
    // 16 rows x 8KB, one byte per 128B line, 1024 lines over 512 threads.
    if (t + 1 < Tv) {
#pragma unroll
      for (int e = 0; e < 2; ++e) {
        int idx = tid + 512 * e;
        int bl = idx >> 6, line = idx & 63;
        const float* p =
            comm + (((long long)(b0 + bl) * Tv + (t + 1)) * Mv) * Hv + line * 32;
        __builtin_prefetch(p, 0, 1);
      }
    }

    // Phase 1: S[m] = Hprev[:,m,:] @ U[m]   (wave m, m<8)
    if (wid < Mv) {
      int m = wid;
      f32x8 c = {};
      for (int kt = 0; kt < 8; ++kt) {
        bf16x16 a = load_a_lds(&Hs[lr][m * Hv + kt * 32]);
        bf16x16 b = load_b_tile(Ub, m * 8 + kt);
        c = wmma_bf16(a, b, c);
      }
#pragma unroll
      for (int i = 0; i < 8; ++i) {
        int pos = (lr < 8) ? lr : lr + 8;  // A-fragment storage position of k=lr
        Sb[m][i + 8 * half][pos] = f2bf(c[i]);
      }
    }
    __syncthreads();

    // Phase 2: H_in = tanh(S@Vr + ip + bias_m)   128 (m,ht) tiles, 8/wave
    for (int it = 0; it < 8; ++it) {
      int idx = it * 16 + wid;
      int m = idx >> 4, ht = idx & 15;
      bf16x16 a = load_a_lds(&Sb[m][lr][0]);
      bf16x16 b = load_b_tile(Vrb, m * 16 + ht);
      f32x8 c = {};
      c = wmma_bf16(a, b, c);
#pragma unroll
      for (int i = 0; i < 8; ++i) {
        int bl = i + 8 * half;
        int h = ht * 16 + lr;
        long long ipa = (((long long)(b0 + bl) * Tv + t) * Mv + m) * Hv + h;
        float v = tanhf(c[i] + comm[ipa] + bias_m[m * Hv + h]);
        Hs[bl][m * Hv + swz32(h)] = f2bf(v);
      }
    }
    __syncthreads();

    // Phase 3: Q,K projections   32 (qk,m,nt) tiles, 2/wave
    for (int it = 0; it < 2; ++it) {
      int idx = it * 16 + wid;
      int qk = idx >> 4; int m = (idx >> 1) & 7; int nt = idx & 1;
      const __bf16* W = qk ? WKb : WQb;
      f32x8 c = {};
      for (int kt = 0; kt < 8; ++kt) {
        bf16x16 a = load_a_lds(&Hs[lr][m * Hv + kt * 32]);
        bf16x16 b = load_b_tile(W, (m * 8 + kt) * 2 + nt);
        c = wmma_bf16(a, b, c);
      }
#pragma unroll
      for (int i = 0; i < 8; ++i) {
        int bl = i + 8 * half;
        float* dst = qk ? &Kb[bl][m][0] : &Qb[bl][m][0];
        dst[nt * 16 + lr] = c[i];
      }
    }
    __syncthreads();

    // Phase 4: 8x8 attention + smoothing (VALU), 1024 entries
    for (int e = 0; e < 2; ++e) {
      int idx = tid + 512 * e;
      int bl = idx >> 6; int m = (idx >> 3) & 7; int n = idx & 7;
      float dot = 0.f;
#pragma unroll
      for (int kk = 0; kk < Kv; ++kk) dot += Qb[bl][m][kk] * Kb[bl][n][kk];
      float raw = 1.f / (1.f + expf(-dot * 0.17677669529663687f));  // /sqrt(32)
      float w = (t == 0) ? raw : 0.5f * (raw + Wb[bl][m][n]);
      Wb[bl][m][n] = w;
      attw[(((long long)t * Bv + (b0 + bl)) * Mv + m) * Mv + n] = w;
    }
    __syncthreads();

    // Phase 5a: Vc[:,n,dtile] for this wave's dtile (dt=wid), kept in registers
    f32x8 vc[8];
    {
      int dt = wid;
#pragma unroll
      for (int n = 0; n < 8; ++n) {
        f32x8 c = {};
        for (int kt = 0; kt < 8; ++kt) {
          bf16x16 a = load_a_lds(&Hs[lr][n * Hv + kt * 32]);
          bf16x16 b = load_b_tile(WVb, (n * 8 + kt) * 16 + dt);
          c = wmma_bf16(a, b, c);
        }
        vc[n] = c;
      }
    }
    __syncthreads();  // all H_in reads finish before H_comm updates

    // Phase 5b: A_comm = w @ Vc; H_comm = H_in + A_comm; emit outputs
    {
      int dt = wid;
#pragma unroll
      for (int i = 0; i < 8; ++i) {
        int bl = i + 8 * half;
        int d = dt * 16 + lr;
#pragma unroll
        for (int m = 0; m < 8; ++m) {
          float ac = 0.f;
#pragma unroll
          for (int n = 0; n < 8; ++n) ac += Wb[bl][m][n] * vc[n][i];
          int hs_idx = m * Hv + swz32(d);
          float hnew = bf2f(Hs[bl][hs_idx]) + ac;
          Hs[bl][hs_idx] = f2bf(hnew);
          hidden[(((long long)t * Bv + (b0 + bl)) * Mv + m) * Hv + d] = hnew;
          comm[(((long long)(b0 + bl) * Tv + t) * Mv + m) * Hv + d] = ac;
        }
      }
    }
    __syncthreads();
  }
}

// ---------- kernel 4: outputs = H_comm @ Wo^T + bo ----------
__global__ __launch_bounds__(32) void k_out(const float* __restrict__ hidden,
                                            const __bf16* __restrict__ Wotb,
                                            const float* __restrict__ bo,
                                            float* __restrict__ out) {
  int rt = blockIdx.x, ct = blockIdx.y;  // 2048 x 8
  int lane = threadIdx.x & 31; int half = lane >> 4; int lr = lane & 15;
  int row = rt * 16 + lr;  // row = t*B + b (hidden layout)
  const float* rowptr = hidden + (long long)row * (Mv * Hv);
  f32x8 c = {};
  for (int kt = 0; kt < 64; ++kt) {
    bf16x16 a  = load_a_f32v(rowptr + kt * 32);
    bf16x16 bb = load_b_tile(Wotb, kt * 8 + ct);
    c = wmma_bf16(a, bb, c);
  }
  float bias = bo[ct * 16 + lr];
#pragma unroll
  for (int i = 0; i < 8; ++i) {
    int rr = rt * 16 + i + 8 * half;
    int t2 = rr >> 7, b2 = rr & 127;
    out[((long long)b2 * Tv + t2) * Ov + ct * 16 + lr] = c[i] + bias;
  }
}

extern "C" void kernel_launch(void* const* d_in, const int* in_sizes, int n_in,
                              void* d_out, int out_size, void* d_ws, size_t ws_size,
                              hipStream_t stream) {
  const float* x      = (const float*)d_in[0];
  const float* Win    = (const float*)d_in[1];
  const float* Wp     = (const float*)d_in[2];
  const float* bp     = (const float*)d_in[3];
  const float* U      = (const float*)d_in[4];
  const float* Vr     = (const float*)d_in[5];
  const float* bias_m = (const float*)d_in[6];
  const float* WQ     = (const float*)d_in[7];
  const float* WK     = (const float*)d_in[8];
  const float* WV     = (const float*)d_in[9];
  const float* Wo     = (const float*)d_in[10];
  const float* bo     = (const float*)d_in[11];

  float* out       = (float*)d_out;
  float* hidden    = out + 4194304LL;         // [T,B,M,H]
  float* attn_in   = hidden + 67108864LL;     // [T,B,M,H]
  float* attn_comm = attn_in + 67108864LL;    // [B,T,M,H] (ip staging, then A_comm)
  float* attn_w    = attn_comm + 67108864LL;  // [1,T,B,M,M]
  __bf16* wsb = (__bf16*)d_ws;

  k_convert<<<(unsigned)((N_TOT + 255) / 256), 256, 0, stream>>>(
      Win, Wp, U, Vr, WQ, WK, WV, Wo, wsb);
  k_ain<<<dim3(2048, 128), 32, 0, stream>>>(x, wsb + O_WIN, attn_in);
  k_ip<<<dim3(2048, 128), 32, 0, stream>>>(attn_in, wsb + O_WP, bp, attn_comm);
  k_recurrent<<<8, 512, 0, stream>>>(bias_m, wsb + O_U, wsb + O_VR, wsb + O_WQ,
                                     wsb + O_WK, wsb + O_WV, hidden, attn_comm,
                                     attn_w);
  k_out<<<dim3(2048, 8), 32, 0, stream>>>(hidden, wsb + O_WOT, bo, out);
}
